// CRF_50689204027649
// MI455X (gfx1250) — matrix operand order repaired and checked by reading, not compiled, not run on hardware
//
#include <hip/hip_runtime.h>
#include <cstdint>
#include <cstddef>

typedef __attribute__((ext_vector_type(16))) _Float16 v16h;
typedef __attribute__((ext_vector_type(2)))  __fp16   fp16x2;
typedef __attribute__((ext_vector_type(8)))  float    v8f;
typedef __attribute__((ext_vector_type(4)))  float    v4f;
typedef __attribute__((ext_vector_type(8)))  uint32_t v8u;
typedef __attribute__((ext_vector_type(4)))  uint32_t v4u;
typedef __attribute__((ext_vector_type(8)))  int32_t  v8i;
typedef __attribute__((ext_vector_type(4)))  int32_t  v4i;

constexpr int S    = 512;
constexpr int T    = 25;
constexpr int WPB  = 8;     // waves per block
constexpr int ROWF = 32;    // floats per row in the LDS tile (TDM zero-pads 25..31)

// ---------------------------------------------------------------------------
// Kernel 1: numerator (labeled-path score) + per-row sequence lengths.
// ---------------------------------------------------------------------------
__global__ void crf_numer_kernel(const float* __restrict__ em,
                                 const float* __restrict__ trans,
                                 const float* __restrict__ startT,
                                 const float* __restrict__ endT,
                                 const int*   __restrict__ labels,
                                 const uint8_t* __restrict__ mask,
                                 float* __restrict__ num,
                                 int*   __restrict__ lenArr,
                                 int Btot)
{
  int b = blockIdx.x * blockDim.x + threadIdx.x;
  if (b >= Btot) return;

  const uint32_t* m4 = (const uint32_t*)(mask + (size_t)b * S);
  int len = 0;
#pragma unroll 4
  for (int s4 = 0; s4 < S / 4; ++s4) {
    uint32_t v = m4[s4];
    len += (int)((v & 0xFFu) + ((v >> 8) & 0xFFu) +
                 ((v >> 16) & 0xFFu) + (v >> 24));
  }

  const int*   lrow = labels + (size_t)b * S;
  const float* erow = em + (size_t)b * S * T;
  int prev = lrow[0];
  float score = startT[prev] + erow[prev];
  // labels are loop-independent loads; unroll so the gathers pipeline
#pragma unroll 4
  for (int s = 1; s < len; ++s) {
    int l = lrow[s];
    score += trans[prev * T + l] + erow[s * T + l];
    prev = l;
  }
  num[b]    = score + endT[prev];
  lenArr[b] = len;
}

// ---------------------------------------------------------------------------
// TDM: DMA one 16-row x 25-col f32 tile (row stride S*T elements) into LDS,
// padded to 32 floats/row.  tile_dim0=32 > tensor_dim0=25 => TDM writes exact
// zeros into LDS columns 25..31 (OOB reads return zero), giving aligned
// 128B rows and implicit zero-padding of the state vector.
// ---------------------------------------------------------------------------
__device__ __forceinline__ void tdm_load_tile(uint64_t gaddr, uint32_t ldsAddr)
{
  v4u g0;
  g0[0] = 1u;                                            // count=1 (valid user D#)
  g0[1] = ldsAddr;                                       // lds_addr[31:0]
  g0[2] = (uint32_t)gaddr;                               // global_addr[31:0]
  g0[3] = ((uint32_t)(gaddr >> 32) & 0x01FFFFFFu)        // global_addr[56:32]
        | (2u << 30);                                    // type = 2 ("image")
  v8i g1;
  g1[0] = (int)(2u << 16);                               // data_size=2 (4B)
  g1[1] = (int)((uint32_t)T << 16);                      // tensor_dim0 = 25
  g1[2] = (int)(16u << 16);                              // tensor_dim1 = 16
  g1[3] = (int)((uint32_t)ROWF << 16);                   // tile_dim0   = 32
  g1[4] = 16;                                            // tile_dim1   = 16
  g1[5] = S * T;                                         // tensor_dim0_stride
  g1[6] = 0;
  g1[7] = 0;
  v4i z4 = {};
#if defined(__clang_major__) && (__clang_major__ >= 23)
  v8i z8 = {};
  __builtin_amdgcn_tensor_load_to_lds(g0, g1, z4, z4, z8, 0);
#else
  __builtin_amdgcn_tensor_load_to_lds(g0, g1, z4, z4, 0);
#endif
}

// ---------------------------------------------------------------------------
// Kernel 2: denominator (forward algorithm) via WMMA + TDM double buffering.
//
//   alpha[b,j] = C_b + log a[b,j]
//   step: m = rowmax(a); p = a/m (f16);  s = E^T x p  (2x wmma f32_16x16x32_f16)
//         a' = s * exp(emit[t]);  commit while t < len_b;  C_b += log m
//
// Fragment geometry puts batch rows in lane columns, so the D->B transpose
// and the row max each need only shfl_xor(...,16) half-wave swaps.
// ---------------------------------------------------------------------------
__global__ void __launch_bounds__(WPB * 32)
crf_denom_kernel(const float* __restrict__ em,
                 const float* __restrict__ trans,
                 const float* __restrict__ startT,
                 const float* __restrict__ endT,
                 const int*   __restrict__ lenArr,
                 float* __restrict__ denArr,
                 int Btot)
{
  __shared__ float smem[WPB][2][16 * ROWF];

  const int lane  = threadIdx.x & 31;
  // launder the wave id into the scalar domain: keeps the whole TDM
  // descriptor address chain in SGPRs (no per-step v_readfirstlane x4)
  const int wave  = __builtin_amdgcn_readfirstlane(threadIdx.x >> 5);
  const int tile  = blockIdx.x * (blockDim.x >> 5) + wave;
  const int baseB = tile * 16;
  if (baseB + 16 > Btot) return;            // wave-uniform: EXEC stays all-1s

  const int c      = lane & 15;
  const int halfhi = lane >> 4;
  const int b      = baseB + c;

  // Constant A operands: EA0 = exp(trans)^T rows j=0..15, EA1 rows j=16..31.
  v16h EA0, EA1;
#pragma unroll
  for (int h = 0; h < 16; ++h) {
    int i  = h + 16 * halfhi;
    int j1 = 16 + c;
    float v0 = (i < T)           ? __expf(trans[i * T + c])  : 0.0f;
    float v1 = (i < T && j1 < T) ? __expf(trans[i * T + j1]) : 0.0f;
    EA0[h] = (_Float16)v0;
    EA1[h] = (_Float16)v1;
  }

  const int len = lenArr[b];

  const uint64_t emBase = (uint64_t)(uintptr_t)em +
                          (uint64_t)baseB * S * T * 4ull;
  const uint32_t lds0 = (uint32_t)(uintptr_t)&smem[wave][0][0];
  const uint32_t lds1 = (uint32_t)(uintptr_t)&smem[wave][1][0];

  // ---- prologue: tile t=0 -> buf0, init state; then issue t=1 -> buf1 ----
  tdm_load_tile(emBase, lds0);
  __builtin_amdgcn_s_wait_tensorcnt(0);
  asm volatile("" ::: "memory");

  float d0[8], d1[8];
  float C = 0.0f;
  {
    const float* row0 = &smem[wave][0][c * ROWF];
    v4f i00 = *(const v4f*)(row0 + 8 * halfhi);
    v4f i01 = *(const v4f*)(row0 + 8 * halfhi + 4);
    v4f i10 = *(const v4f*)(row0 + 16 + 8 * halfhi);
    v4f i11 = *(const v4f*)(row0 + 16 + 8 * halfhi + 4);
#pragma unroll
    for (int r = 0; r < 4; ++r) {
      int j0 = r + 8 * halfhi;
      d0[r] = __expf(startT[j0] + i00[r]);
      int j0b = 4 + r + 8 * halfhi;
      d0[4 + r] = __expf(startT[j0b] + i01[r]);
      int j1 = 16 + r + 8 * halfhi;
      int j1c = (j1 < T) ? j1 : (T - 1);
      float va = __expf(startT[j1c] + i10[r]);
      d1[r] = (j1 < T) ? va : 0.0f;
      int j1b = 20 + r + 8 * halfhi;
      int j1bc = (j1b < T) ? j1b : (T - 1);
      float vb = __expf(startT[j1bc] + i11[r]);
      d1[4 + r] = (j1b < T) ? vb : 0.0f;
    }
  }
  tdm_load_tile(emBase + (uint64_t)T * 4ull, lds1);

  const v8f zacc = {};

  for (int t = 1; t < S; ++t) {
    // issue next tile into the other buffer before waiting on the current one
    int nxt = (t + 1 < S) ? (t + 1) : (S - 1);     // redundant tail re-load: harmless
    tdm_load_tile(emBase + (uint64_t)nxt * T * 4ull, (t & 1) ? lds0 : lds1);
    __builtin_amdgcn_s_wait_tensorcnt(1);           // in-order: tile t is complete
    asm volatile("" ::: "memory");

    const float* rowp = &smem[wave][t & 1][c * ROWF];
    v4f e00 = *(const v4f*)(rowp + 8 * halfhi);
    v4f e01 = *(const v4f*)(rowp + 8 * halfhi + 4);
    v4f e10 = *(const v4f*)(rowp + 16 + 8 * halfhi);
    v4f e11 = *(const v4f*)(rowp + 16 + 8 * halfhi + 4);

    // per-batch-row max over j (padded entries are exact zeros; a >= 0)
    float mx = d0[0];
#pragma unroll
    for (int r = 1; r < 8; ++r) mx = fmaxf(mx, d0[r]);
#pragma unroll
    for (int r = 0; r < 8; ++r) mx = fmaxf(mx, d1[r]);
    mx = fmaxf(mx, __shfl_xor(mx, 16, 32));
    float rinv = __builtin_amdgcn_rcpf(mx);

    // Build B operand p = a/m (f16).  Each lane only needs ONE dword from its
    // half-wave partner per k: low lanes need the partner's q0, high lanes the
    // partner's q1.  Pre-select what to send, so the transpose costs just
    // 4 lane-swaps total (instead of 8).
    v8u pvu;
#pragma unroll
    for (int k = 0; k < 4; ++k) {
      fp16x2 p0 = __builtin_amdgcn_cvt_pkrtz(d0[2 * k] * rinv, d0[2 * k + 1] * rinv);
      fp16x2 p1 = __builtin_amdgcn_cvt_pkrtz(d1[2 * k] * rinv, d1[2 * k + 1] * rinv);
      uint32_t q0 = __builtin_bit_cast(uint32_t, p0);
      uint32_t q1 = __builtin_bit_cast(uint32_t, p1);
      uint32_t z  = halfhi ? q0 : q1;                    // dword the partner needs
      uint32_t w  = (uint32_t)__shfl_xor((int)z, 16, 32);
      pvu[k]     = halfhi ? w  : q0;   // i-halves 0..7  (lane<16) / 16..23
      pvu[k + 4] = halfhi ? q1 : w;    // i-halves 8..15 (lane<16) / 24..31
    }
    v16h pv = __builtin_bit_cast(v16h, pvu);

    v8f s0 = __builtin_amdgcn_wmma_f32_16x16x32_f16(
        false, EA0, false, pv, (short)0, zacc, false, false);
    v8f s1 = __builtin_amdgcn_wmma_f32_16x16x32_f16(
        false, EA1, false, pv, (short)0, zacc, false, false);

    const bool  commit = (t < len);
    const float logm   = __logf(mx);
    // a' = s * exp(emit);  padded j: s==0 and emit==0 -> a'==0 stays exact.
#pragma unroll
    for (int r = 0; r < 4; ++r) {
      float a0 = s0[r]     * __expf(e00[r]);
      float a1 = s0[4 + r] * __expf(e01[r]);
      float a2 = s1[r]     * __expf(e10[r]);
      float a3 = s1[4 + r] * __expf(e11[r]);
      d0[r]     = commit ? a0 : d0[r];
      d0[4 + r] = commit ? a1 : d0[4 + r];
      d1[r]     = commit ? a2 : d1[r];
      d1[4 + r] = commit ? a3 : d1[4 + r];
    }
    C = commit ? (C + logm) : C;
  }

  // denominator_b = C_b + log( sum_j a[b,j] * exp(end[j]) )
  // (padded d1 entries are exactly 0, so clamped end-index reads are safe)
  float sum = 0.0f;
#pragma unroll
  for (int r = 0; r < 8; ++r) {
    int j0 = r + 8 * halfhi;
    sum += d0[r] * __expf(endT[j0]);
    int j1 = 16 + r + 8 * halfhi;
    int j1c = (j1 < T) ? j1 : (T - 1);
    sum += d1[r] * __expf(endT[j1c]);
  }
  sum += __shfl_xor(sum, 16, 32);
  if (!halfhi) denArr[b] = C + __logf(sum);
}

// ---------------------------------------------------------------------------
// Kernel 3: deterministic final reduction -> out[0] = sum(num-den)/sum(len)
// ---------------------------------------------------------------------------
__global__ void crf_reduce_kernel(const float* __restrict__ num,
                                  const float* __restrict__ den,
                                  const int*   __restrict__ lenArr,
                                  float* __restrict__ out, int Btot)
{
  __shared__ float sL[256];
  __shared__ float sN[256];
  int tid = threadIdx.x;
  float accL = 0.0f, accN = 0.0f;
  for (int i = tid; i < Btot; i += 256) {
    accL += num[i] - den[i];
    accN += (float)lenArr[i];
  }
  sL[tid] = accL; sN[tid] = accN;
  __syncthreads();
  for (int s = 128; s > 0; s >>= 1) {
    if (tid < s) { sL[tid] += sL[tid + s]; sN[tid] += sN[tid + s]; }
    __syncthreads();
  }
  if (tid == 0) out[0] = sL[0] / sN[0];
}

// ---------------------------------------------------------------------------
extern "C" void kernel_launch(void* const* d_in, const int* in_sizes, int n_in,
                              void* d_out, int out_size, void* d_ws, size_t ws_size,
                              hipStream_t stream)
{
  const float*   em    = (const float*)d_in[0];
  const float*   trans = (const float*)d_in[1];
  const float*   st    = (const float*)d_in[2];
  const float*   en    = (const float*)d_in[3];
  const int*     lab   = (const int*)d_in[4];
  const uint8_t* mask  = (const uint8_t*)d_in[5];
  const int Btot = in_sizes[0] / (S * T);   // 2048

  float* num  = (float*)d_ws;
  float* den  = num + Btot;
  int*   lenA = (int*)(den + Btot);
  float* out  = (float*)d_out;

  crf_numer_kernel<<<(Btot + 255) / 256, 256, 0, stream>>>(
      em, trans, st, en, lab, mask, num, lenA, Btot);

  const int tiles  = Btot / 16;             // 16 batch rows per wave
  const int blocks = (tiles + WPB - 1) / WPB;
  crf_denom_kernel<<<blocks, WPB * 32, 0, stream>>>(
      em, trans, st, en, lenA, den, Btot);

  crf_reduce_kernel<<<1, 256, 0, stream>>>(num, den, lenA, out, Btot);
}
